// Decoder_GMM_16432544875083
// MI455X (gfx1250) — compile-verified, matrix operand order checked by reading, not run on hardware
//
#include <hip/hip_runtime.h>
#include <math.h>

// ---------------- problem dims ----------------
#define B_     64
#define T_ENC_ 200
#define T_DEC_ 400
#define MEL_   80
#define R_     2
#define STEPS_ 200
#define ENC_   512
#define PRE_   256
#define ATTN_  128
#define ARNN_  1024
#define DRNN_  1024
#define GATES_ 4096
#define FILT_  32
#define KSZ_   31
#define KA_    1792   // PRE + ENC + ARNN  (attention-LSTM concat input)
#define KD_    2560   // ARNN + ENC + DRNN (decoder-LSTM concat input)

typedef __attribute__((ext_vector_type(16))) __bf16 v16bf;
typedef __attribute__((ext_vector_type(8)))  __bf16 v8bf;
typedef __attribute__((ext_vector_type(8)))  float  v8f;

__device__ __forceinline__ float sigf(float x) { return 1.f / (1.f + __expf(-x)); }

// ----------------------------------------------------------------------------
// Pack W = [W1 | W2] (concat along K) of shape (N x K) fp32 into bf16 tiles in
// the CDNA5 WMMA B-operand (32x16 KxN) lane layout:
//   element index = ((nt*KT + kt)*32 + lane)*16 + slot
//   lane: n = nt*16 + (lane&15); kbase = (lane>=16 ? 8 : 0)
//   slot s: k = kt*32 + kbase + (s<8 ? s : s+8)
// Each wave-tile is 1KB contiguous -> coalesced per-step streaming from L2.
// ----------------------------------------------------------------------------
__global__ void pack_weight_bf16(const float* __restrict__ W1, int K1,
                                 const float* __restrict__ W2, int K2,
                                 __bf16* __restrict__ dst, int N, int K)
{
    const int KT = K >> 5;
    const size_t total = (size_t)(N >> 4) * KT * 512;
    for (size_t idx = (size_t)blockIdx.x * blockDim.x + threadIdx.x; idx < total;
         idx += (size_t)gridDim.x * blockDim.x) {
        const int s  = (int)(idx & 15);
        const int l  = (int)((idx >> 4) & 31);
        const size_t tile = idx >> 9;
        const int kt = (int)(tile % KT);
        const int nt = (int)(tile / KT);
        const int n  = nt * 16 + (l & 15);
        const int k  = kt * 32 + ((l >> 4) * 8) + (s < 8 ? s : s + 8);
        float w;
        if (k < K1) w = W1[(size_t)n * K1 + k];
        else        w = W2[(size_t)n * K2 + (k - K1)];
        dst[idx] = (__bf16)w;
    }
}

__global__ void f32_to_bf16(const float* __restrict__ src, __bf16* __restrict__ dst, size_t n)
{
    for (size_t i = (size_t)blockIdx.x * blockDim.x + threadIdx.x; i < n;
         i += (size_t)gridDim.x * blockDim.x)
        dst[i] = (__bf16)src[i];
}

__global__ void zero_f32(float* __restrict__ p, size_t n)
{
    for (size_t i = (size_t)blockIdx.x * blockDim.x + threadIdx.x; i < n;
         i += (size_t)gridDim.x * blockDim.x) p[i] = 0.f;
}

__global__ void zero_bf16(__bf16* __restrict__ p, size_t n)
{
    for (size_t i = (size_t)blockIdx.x * blockDim.x + threadIdx.x; i < n;
         i += (size_t)gridDim.x * blockDim.x) p[i] = (__bf16)0.f;
}

// ----------------------------------------------------------------------------
// Generic C(MxN fp32) = A(MxK bf16) x Bpacked via v_wmma_f32_16x16x32_bf16.
// Used once per launch for pm = enc @ W_mem^T.
// ----------------------------------------------------------------------------
__global__ void __launch_bounds__(128)
gemm_bf16_wmma(const __bf16* __restrict__ A, const __bf16* __restrict__ Bp,
               float* __restrict__ C, int M, int N, int K)
{
    const int wave = threadIdx.x >> 5;
    const int lane = threadIdx.x & 31;
    const int nt = blockIdx.x * 4 + wave;
    const int mt = blockIdx.y;
    const int KT = K >> 5;
    const int m  = mt * 16 + (lane & 15);
    const int khalf = (lane >> 4) * 8;

    v8f acc = {};
    const __bf16* arow = A + (size_t)m * K + khalf;
    const __bf16* bptr = Bp + ((size_t)nt * KT * 32 + lane) * 16;

    for (int kt = 0; kt < KT; ++kt) {
        v8bf alo = *(const v8bf*)(arow);
        v8bf ahi = *(const v8bf*)(arow + 16);
        v16bf a;
#pragma unroll
        for (int i = 0; i < 8; ++i) { a[i] = alo[i]; a[i + 8] = ahi[i]; }
        v16bf b = *(const v16bf*)(bptr);
        acc = __builtin_amdgcn_wmma_f32_16x16x32_bf16(false, a, false, b,
                                                      (short)0, acc, false, false);
        arow += 32;
        bptr += 512;
    }
    const int col   = nt * 16 + (lane & 15);
    const int rbase = mt * 16 + ((lane >> 4) ? 8 : 0);
#pragma unroll
    for (int i = 0; i < 8; ++i)
        C[(size_t)(rbase + i) * N + col] = acc[i];
}

// ----------------------------------------------------------------------------
// Fused LSTM cell: gates GEMM + pointwise update, one kernel.
// A wave owns unit columns u = nt*16..nt*16+15 for batch rows mt*16..mt*16+15
// and accumulates ALL FOUR gate tiles (i,f,g,o at N-tiles nt, nt+H/16, ...)
// so the sigmoid/tanh cell update runs in registers right after the K-loop.
// A is loaded once per kt and feeds 4 WMMAs (4x A reuse, 2.5 loads/WMMA).
// Grid: (H/64, B/16), 128 threads (4 waves).
// ----------------------------------------------------------------------------
__global__ void __launch_bounds__(128)
lstm_fused_wmma(const __bf16* __restrict__ A, const __bf16* __restrict__ Bp,
                const float* __restrict__ bih, const float* __restrict__ bhh,
                float* __restrict__ h, float* __restrict__ c,
                __bf16* __restrict__ hbf1, __bf16* __restrict__ hbf2,
                int H, int K, int xstride1, int xoff1, int xstride2, int xoff2)
{
    const int wave = threadIdx.x >> 5;
    const int lane = threadIdx.x & 31;
    const int nt = blockIdx.x * 4 + wave;   // 0 .. H/16-1 : unit-column tile
    const int mt = blockIdx.y;              // 0 .. B/16-1 : batch tile
    const int KT = K >> 5;
    const int ng = H >> 4;                  // N-tiles per gate
    const int m  = mt * 16 + (lane & 15);
    const int khalf = (lane >> 4) * 8;

    v8f accI = {}, accF = {}, accG = {}, accO = {};
    const __bf16* arow = A + (size_t)m * K + khalf;
    const size_t bstep = (size_t)KT * 32 * 16;
    const __bf16* bI = Bp + (size_t)nt * bstep + (size_t)lane * 16;
    const __bf16* bF = bI + (size_t)ng * bstep;
    const __bf16* bG = bF + (size_t)ng * bstep;
    const __bf16* bO = bG + (size_t)ng * bstep;

    for (int kt = 0; kt < KT; ++kt) {
        v8bf alo = *(const v8bf*)(arow);
        v8bf ahi = *(const v8bf*)(arow + 16);
        v16bf a;
#pragma unroll
        for (int i = 0; i < 8; ++i) { a[i] = alo[i]; a[i + 8] = ahi[i]; }
        v16bf tI = *(const v16bf*)(bI);
        v16bf tF = *(const v16bf*)(bF);
        v16bf tG = *(const v16bf*)(bG);
        v16bf tO = *(const v16bf*)(bO);
        accI = __builtin_amdgcn_wmma_f32_16x16x32_bf16(false, a, false, tI,
                                                       (short)0, accI, false, false);
        accF = __builtin_amdgcn_wmma_f32_16x16x32_bf16(false, a, false, tF,
                                                       (short)0, accF, false, false);
        accG = __builtin_amdgcn_wmma_f32_16x16x32_bf16(false, a, false, tG,
                                                       (short)0, accG, false, false);
        accO = __builtin_amdgcn_wmma_f32_16x16x32_bf16(false, a, false, tO,
                                                       (short)0, accO, false, false);
        arow += 32;
        bI += 512; bF += 512; bG += 512; bO += 512;
    }

    // epilogue: LSTM cell update in registers (torch gate order i,f,g,o)
    const int u     = nt * 16 + (lane & 15);
    const int rbase = mt * 16 + ((lane >> 4) ? 8 : 0);
    const float bi = bih[u]         + bhh[u];
    const float bf = bih[H + u]     + bhh[H + u];
    const float bg = bih[2 * H + u] + bhh[2 * H + u];
    const float bo = bih[3 * H + u] + bhh[3 * H + u];
#pragma unroll
    for (int i = 0; i < 8; ++i) {
        const int b = rbase + i;
        const size_t idx = (size_t)b * H + u;
        const float c2 = sigf(accF[i] + bf) * c[idx]
                       + sigf(accI[i] + bi) * tanhf(accG[i] + bg);
        const float hn = sigf(accO[i] + bo) * tanhf(c2);
        c[idx] = c2;
        h[idx] = hn;
        hbf1[(size_t)b * xstride1 + xoff1 + u] = (__bf16)hn;
        if (hbf2) hbf2[(size_t)b * xstride2 + xoff2 + u] = (__bf16)hn;
    }
}

// ---------------- prenet: p = relu(relu(x W1^T + b1) W2^T + b2) -------------
__global__ void __launch_bounds__(256)
prenet_kernel(const float* __restrict__ inputs, int step,
              const float* __restrict__ W1, const float* __restrict__ b1,
              const float* __restrict__ W2, const float* __restrict__ b2,
              __bf16* __restrict__ xa)
{
    __shared__ float xs[MEL_];
    __shared__ float h1[PRE_];
    const int b = blockIdx.x, tid = threadIdx.x;
    if (tid < MEL_)
        xs[tid] = (step == 0) ? 0.f
                 : inputs[((size_t)b * T_DEC_ + (step * R_ - 1)) * MEL_ + tid];
    __syncthreads();
    float acc = b1[tid];
    const float* w = W1 + (size_t)tid * MEL_;
    for (int k = 0; k < MEL_; ++k) acc += w[k] * xs[k];
    h1[tid] = fmaxf(acc, 0.f);
    __syncthreads();
    float acc2 = b2[tid];
    const float* w2 = W2 + (size_t)tid * PRE_;
    for (int k = 0; k < PRE_; ++k) acc2 += w2[k] * h1[k];
    xa[(size_t)b * KA_ + tid] = (__bf16)fmaxf(acc2, 0.f);
}

// ------- location-sensitive attention: conv + energy + softmax + context ----
__global__ void __launch_bounds__(256)
attention_kernel(const float* __restrict__ ah,
                 const float* __restrict__ Wq,
                 const float* __restrict__ Wconv,
                 const float* __restrict__ Wloc,
                 const float* __restrict__ vvec,
                 const float* __restrict__ pm,
                 const float* __restrict__ enc,
                 float* __restrict__ aw,
                 float* __restrict__ ctx,
                 __bf16* __restrict__ xa, __bf16* __restrict__ xd,
                 float* __restrict__ attn_out, int step)
{
    __shared__ float aw_s[T_ENC_];
    __shared__ float q_s[ATTN_];
    __shared__ float wloc_s[ATTN_ * FILT_];
    __shared__ float wconv_s[FILT_ * KSZ_];
    __shared__ float v_s[ATTN_];
    __shared__ float e_s[T_ENC_];
    __shared__ float red_s[256];
    const int b = blockIdx.x, tid = threadIdx.x;

    for (int t = tid; t < T_ENC_; t += 256) aw_s[t] = aw[(size_t)b * T_ENC_ + t];
    for (int i = tid; i < ATTN_ * FILT_; i += 256) wloc_s[i] = Wloc[i];
    for (int i = tid; i < FILT_ * KSZ_; i += 256) wconv_s[i] = Wconv[i];
    if (tid < ATTN_) {
        v_s[tid] = vvec[tid];
        float acc = 0.f;
        const float* w  = Wq + (size_t)tid * ARNN_;
        const float* hh = ah + (size_t)b * ARNN_;
        for (int k = 0; k < ARNN_; ++k) acc += w[k] * hh[k];
        q_s[tid] = acc;
    }
    __syncthreads();

    const int pad = (KSZ_ - 1) / 2;
    for (int t = tid; t < T_ENC_; t += 256) {
        float loc[FILT_];
#pragma unroll
        for (int f = 0; f < FILT_; ++f) loc[f] = 0.f;
        for (int k = 0; k < KSZ_; ++k) {
            const int tt = t + k - pad;
            if (tt >= 0 && tt < T_ENC_) {
                const float a = aw_s[tt];
#pragma unroll
                for (int f = 0; f < FILT_; ++f) loc[f] += wconv_s[f * KSZ_ + k] * a;
            }
        }
        const float* pmrow = pm + ((size_t)b * T_ENC_ + t) * ATTN_;
        float e = 0.f;
        for (int a0 = 0; a0 < ATTN_; ++a0) {
            float s = q_s[a0] + pmrow[a0];
            const float* wl = wloc_s + a0 * FILT_;
#pragma unroll
            for (int f = 0; f < FILT_; ++f) s += wl[f] * loc[f];
            e += v_s[a0] * tanhf(s);
        }
        e_s[t] = e;
    }
    __syncthreads();

    // parallel max reduction
    red_s[tid] = (tid < T_ENC_) ? e_s[tid] : -1e30f;
    __syncthreads();
    for (int off = 128; off > 0; off >>= 1) {
        if (tid < off) red_s[tid] = fmaxf(red_s[tid], red_s[tid + off]);
        __syncthreads();
    }
    const float mx = red_s[0];
    __syncthreads();
    // parallel sum reduction of exp(e - mx)
    red_s[tid] = (tid < T_ENC_) ? __expf(e_s[tid] - mx) : 0.f;
    __syncthreads();
    for (int off = 128; off > 0; off >>= 1) {
        if (tid < off) red_s[tid] += red_s[tid + off];
        __syncthreads();
    }
    const float inv = 1.f / red_s[0];
    __syncthreads();

    for (int t = tid; t < T_ENC_; t += 256) {
        const float al = __expf(e_s[t] - mx) * inv;
        e_s[t] = al;                                           // reuse as align
        aw[(size_t)b * T_ENC_ + t] = al;                       // carry
        attn_out[((size_t)b * STEPS_ + step) * T_ENC_ + t] = al;
    }
    __syncthreads();

    for (int d = tid; d < ENC_; d += 256) {
        float acc = 0.f;
        const float* er = enc + (size_t)b * T_ENC_ * ENC_ + d;
        for (int t = 0; t < T_ENC_; ++t) acc += e_s[t] * er[(size_t)t * ENC_];
        ctx[(size_t)b * ENC_ + d] = acc;
        const __bf16 cb = (__bf16)acc;
        xa[(size_t)b * KA_ + PRE_ + d]  = cb;
        xd[(size_t)b * KD_ + ARNN_ + d] = cb;
    }
}

// ---------------- output projections (mel + stop) ---------------------------
__global__ void __launch_bounds__(192)
proj_kernel(const float* __restrict__ dh, const float* __restrict__ ctx,
            const float* __restrict__ Wmel, const float* __restrict__ bmel,
            const float* __restrict__ Wstop, const float* __restrict__ bstop,
            float* __restrict__ mel_out, float* __restrict__ stop_out, int step)
{
    const int b = blockIdx.x, tid = threadIdx.x;
    const float* h  = dh  + (size_t)b * DRNN_;
    const float* cc = ctx + (size_t)b * ENC_;
    if (tid < MEL_ * R_) {
        const float* w = Wmel + (size_t)tid * (DRNN_ + ENC_);
        float acc = bmel[tid];
        for (int k = 0; k < DRNN_; ++k) acc += w[k] * h[k];
        for (int k = 0; k < ENC_; ++k)  acc += w[DRNN_ + k] * cc[k];
        const int r = tid / MEL_, m = tid - r * MEL_;
        mel_out[((size_t)b * T_DEC_ + step * R_ + r) * MEL_ + m] = acc;
    } else if (tid == MEL_ * R_) {
        float acc = bstop[0];
        for (int k = 0; k < DRNN_; ++k) acc += Wstop[k] * h[k];
        for (int k = 0; k < ENC_; ++k)  acc += Wstop[DRNN_ + k] * cc[k];
        const float s = sigf(acc);
        stop_out[(size_t)b * T_DEC_ + step * R_]     = s;
        stop_out[(size_t)b * T_DEC_ + step * R_ + 1] = s;
    }
}

// ----------------------------------------------------------------------------
extern "C" void kernel_launch(void* const* d_in, const int* in_sizes, int n_in,
                              void* d_out, int out_size, void* d_ws, size_t ws_size,
                              hipStream_t stream)
{
    (void)in_sizes; (void)n_in; (void)out_size; (void)ws_size;
    const float* enc    = (const float*)d_in[0];
    const float* inputs = (const float*)d_in[1];
    const float* W_pre1 = (const float*)d_in[2];
    const float* b_pre1 = (const float*)d_in[3];
    const float* W_pre2 = (const float*)d_in[4];
    const float* b_pre2 = (const float*)d_in[5];
    const float* W_mem  = (const float*)d_in[6];
    const float* Wih_a  = (const float*)d_in[7];
    const float* Whh_a  = (const float*)d_in[8];
    const float* bih_a  = (const float*)d_in[9];
    const float* bhh_a  = (const float*)d_in[10];
    const float* W_q    = (const float*)d_in[11];
    const float* W_conv = (const float*)d_in[12];
    const float* W_loc  = (const float*)d_in[13];
    const float* vvec   = (const float*)d_in[14];
    const float* Wih_d  = (const float*)d_in[15];
    const float* Whh_d  = (const float*)d_in[16];
    const float* bih_d  = (const float*)d_in[17];
    const float* bhh_d  = (const float*)d_in[18];
    const float* W_mel  = (const float*)d_in[19];
    const float* b_mel  = (const float*)d_in[20];
    const float* W_stop = (const float*)d_in[21];
    const float* b_stop = (const float*)d_in[22];

    float* mel_out  = (float*)d_out;
    float* stop_out = mel_out + (size_t)B_ * T_DEC_ * MEL_;
    float* attn_out = stop_out + (size_t)B_ * T_DEC_;

    // ---- workspace carve-out (~58MB) ----
    char* ws = (char*)d_ws;
    size_t off = 0;
    auto carve = [&](size_t bytes) -> char* {
        char* p = ws + off;
        off = (off + bytes + 255) & ~(size_t)255;
        return p;
    };
    __bf16* Wa_pk   = (__bf16*)carve((size_t)GATES_ * KA_ * 2);
    __bf16* Wd_pk   = (__bf16*)carve((size_t)GATES_ * KD_ * 2);
    __bf16* Wmem_pk = (__bf16*)carve((size_t)ATTN_ * ENC_ * 2);
    __bf16* enc_bf  = (__bf16*)carve((size_t)B_ * T_ENC_ * ENC_ * 2);
    float*  pm      = (float*) carve((size_t)B_ * T_ENC_ * ATTN_ * 4);
    __bf16* xa      = (__bf16*)carve((size_t)B_ * KA_ * 2);
    __bf16* xd      = (__bf16*)carve((size_t)B_ * KD_ * 2);
    const size_t stateN = (size_t)B_ * (ARNN_ + ARNN_ + DRNN_ + DRNN_ + ENC_ + T_ENC_);
    float* state = (float*)carve(stateN * 4);
    float* ah = state;
    float* ac = ah + (size_t)B_ * ARNN_;
    float* dh = ac + (size_t)B_ * ARNN_;
    float* dc = dh + (size_t)B_ * DRNN_;
    float* ctx = dc + (size_t)B_ * DRNN_;
    float* aw  = ctx + (size_t)B_ * ENC_;

    // ---- one-time per launch: init state, pack weights, precompute pm ----
    zero_f32 <<<512, 256, 0, stream>>>(state, stateN);
    zero_bf16<<<256, 256, 0, stream>>>(xa, (size_t)B_ * KA_);
    zero_bf16<<<256, 256, 0, stream>>>(xd, (size_t)B_ * KD_);

    pack_weight_bf16<<<1024, 256, 0, stream>>>(Wih_a, PRE_ + ENC_, Whh_a, ARNN_,
                                               Wa_pk, GATES_, KA_);
    pack_weight_bf16<<<1024, 256, 0, stream>>>(Wih_d, ARNN_ + ENC_, Whh_d, DRNN_,
                                               Wd_pk, GATES_, KD_);
    pack_weight_bf16<<<128, 256, 0, stream>>>(W_mem, ENC_, W_mem, ENC_,
                                              Wmem_pk, ATTN_, ENC_);
    f32_to_bf16<<<2048, 256, 0, stream>>>(enc, enc_bf, (size_t)B_ * T_ENC_ * ENC_);

    // pm = enc @ W_mem^T : M = 12800, N = 128, K = 512
    gemm_bf16_wmma<<<dim3(ATTN_ / 64, (B_ * T_ENC_) / 16), 128, 0, stream>>>(
        enc_bf, Wmem_pk, pm, B_ * T_ENC_, ATTN_, ENC_);

    // ---- 200 sequential decoder steps (5 kernels/step) ----
    const dim3 glstm(ARNN_ / 64, B_ / 16);  // (16, 4), 4 waves/block
    for (int s = 0; s < STEPS_; ++s) {
        prenet_kernel<<<B_, 256, 0, stream>>>(inputs, s, W_pre1, b_pre1,
                                              W_pre2, b_pre2, xa);
        // attention LSTM: gates GEMM (K=1792) + cell update fused
        lstm_fused_wmma<<<glstm, 128, 0, stream>>>(
            xa, Wa_pk, bih_a, bhh_a, ah, ac,
            xa, xd, ARNN_, KA_, KA_, PRE_ + ENC_, KD_, 0);
        attention_kernel<<<B_, 256, 0, stream>>>(ah, W_q, W_conv, W_loc, vvec,
                                                 pm, enc, aw, ctx, xa, xd,
                                                 attn_out, s);
        // decoder LSTM: gates GEMM (K=2560) + cell update fused
        lstm_fused_wmma<<<glstm, 128, 0, stream>>>(
            xd, Wd_pk, bih_d, bhh_d, dh, dc,
            xd, (__bf16*)nullptr, DRNN_, KD_, KD_, ARNN_ + ENC_, 0, 0);
        proj_kernel<<<B_, 192, 0, stream>>>(dh, ctx, W_mel, b_mel,
                                            W_stop, b_stop,
                                            mel_out, stop_out, s);
    }
}
